// CrossAttention_36129264894376
// MI455X (gfx1250) — matrix operand order repaired and checked by reading, not compiled, not run on hardware
//
#include <hip/hip_runtime.h>
#include <hip/hip_bf16.h>

#define DIM   1024
#define HEADS 16
#define DHEAD 64
#define BATCH 2
#define SEQ   2048
#define MROWS (BATCH * SEQ)          // 4096 token rows
#define SCALE 0.03125f               // DIM^-0.5 = 1/32

typedef __attribute__((ext_vector_type(16))) _Float16 v16h;
typedef __attribute__((ext_vector_type(8)))  _Float16 h8;
typedef __attribute__((ext_vector_type(8)))  float    v8f;

__device__ __forceinline__ v16h combine16(h8 lo, h8 hi) {
    return __builtin_shufflevector(lo, hi, 0,1,2,3,4,5,6,7,8,9,10,11,12,13,14,15);
}

// Convert 8 contiguous f32 -> h8 (two b128 loads + cvt)
__device__ __forceinline__ h8 cvt8(const float* __restrict__ p) {
    float4 a = *(const float4*)p;
    float4 b = *(const float4*)(p + 4);
    h8 r;
    r[0]=(_Float16)a.x; r[1]=(_Float16)a.y; r[2]=(_Float16)a.z; r[3]=(_Float16)a.w;
    r[4]=(_Float16)b.x; r[5]=(_Float16)b.y; r[6]=(_Float16)b.z; r[7]=(_Float16)b.w;
    return r;
}

// One 16-byte async global->LDS transfer (ASYNCcnt-tracked).
// lds_off is the raw LDS byte offset (low 32 bits of the generic pointer).
__device__ __forceinline__ void async_copy_b128(unsigned lds_off, const void* gp) {
    asm volatile("global_load_async_to_lds_b128 %0, %1, off"
                 :: "v"(lds_off), "v"(gp) : "memory");
}

// ---------------------------------------------------------------------------
// Kernel 1: f32 -> f16 weight conversion (Wq|Wk|Wv concatenated in ws)
// ---------------------------------------------------------------------------
__global__ __launch_bounds__(256) void cvt_w_kernel(
    const float* __restrict__ Wq, const float* __restrict__ Wk,
    const float* __restrict__ Wv, _Float16* __restrict__ out) {
    int idx = blockIdx.x * blockDim.x + threadIdx.x;   // < 3*DIM*DIM
    const float* src = (idx < DIM*DIM) ? Wq : (idx < 2*DIM*DIM ? Wk : Wv);
    int off = idx & (DIM*DIM - 1);
    out[idx] = (_Float16)src[off];
}

// ---------------------------------------------------------------------------
// Kernel 2: projection GEMM  Y = X * W^T + b   (M=4096, N=1024, K=1024)
// 1 wave -> 32x32 output tile (2x2 v_wmma_f32_16x16x32_f16 accumulators).
// z=0 -> Qh[b,h,n,d], z=1 -> Kh[b,h,n,d], z=2 -> Vt[b,h,d,n]  (all f16)
// ---------------------------------------------------------------------------
__global__ __launch_bounds__(128) void proj_kernel(
    const float* __restrict__ Xq, const float* __restrict__ Xk,
    const float* __restrict__ Xv,
    const _Float16* __restrict__ Wqh, const _Float16* __restrict__ Wkh,
    const _Float16* __restrict__ Wvh,
    const float* __restrict__ bq, const float* __restrict__ bk,
    const float* __restrict__ bv,
    _Float16* __restrict__ Qh, _Float16* __restrict__ Kh,
    _Float16* __restrict__ Vt) {

    const int z = blockIdx.z;
    const float*    X    = (z == 0) ? Xq  : (z == 1) ? Xk  : Xv;
    const _Float16* W    = (z == 0) ? Wqh : (z == 1) ? Wkh : Wvh;
    const float*    bias = (z == 0) ? bq  : (z == 1) ? bk  : bv;

    const int lane = threadIdx.x & 31;
    const int wave = threadIdx.x >> 5;
    const int lm   = lane & 15;           // row/col within 16
    const int half = lane >> 4;           // lane-half selects contiguous K half
    const int m0 = blockIdx.x * 64 + (wave >> 1) * 32;
    const int n0 = blockIdx.y * 64 + (wave &  1) * 32;

    v8f acc[2][2] = {};

    for (int k0 = 0; k0 < DIM; k0 += 32) {
        // A tiles (16x32 f16, converted from f32 on the fly)
        v16h a[2];
#pragma unroll
        for (int mi = 0; mi < 2; ++mi) {
            const float* xrow = X + (m0 + mi*16 + lm) * DIM + k0;
            h8 lo = cvt8(xrow + half * 16);
            h8 hi = cvt8(xrow + half * 16 + 8);
            a[mi] = combine16(lo, hi);
        }
        // B tiles (32x16 f16): lane = output col, 16 contiguous K per lane-half
        v16h bmat[2];
#pragma unroll
        for (int ni = 0; ni < 2; ++ni) {
            const _Float16* wrow = W + (n0 + ni*16 + lm) * DIM + k0 + half * 16;
            bmat[ni] = *(const v16h*)wrow;
        }
        if (k0 + 64 < DIM) {
            __builtin_prefetch(X + (m0 + lm) * DIM + k0 + 64, 0, 3);
            __builtin_prefetch(W + (n0 + lm) * DIM + k0 + 64, 0, 3);
        }
#pragma unroll
        for (int mi = 0; mi < 2; ++mi)
#pragma unroll
            for (int ni = 0; ni < 2; ++ni)
                acc[mi][ni] = __builtin_amdgcn_wmma_f32_16x16x32_f16(
                    false, a[mi], false, bmat[ni], (short)0, acc[mi][ni],
                    false, false);
    }

    // Store with bias, scattered into attention-friendly layouts.
#pragma unroll
    for (int ni = 0; ni < 2; ++ni) {
        const int col = n0 + ni*16 + lm;
        const float bn = bias[col];
        const int hh = col >> 6;          // head
        const int dd = col & 63;          // dim-in-head
#pragma unroll
        for (int mi = 0; mi < 2; ++mi) {
#pragma unroll
            for (int i = 0; i < 8; ++i) {
                const int m   = m0 + mi*16 + half*8 + i;
                const int bb  = m >> 11;          // / SEQ
                const int tok = m & (SEQ - 1);
                const _Float16 hv = (_Float16)(acc[mi][ni][i] + bn);
                if (z == 0)
                    Qh[((bb*HEADS + hh)*SEQ + tok)*DHEAD + dd] = hv;
                else if (z == 1)
                    Kh[((bb*HEADS + hh)*SEQ + tok)*DHEAD + dd] = hv;
                else
                    Vt[((bb*HEADS + hh)*DHEAD + dd)*SEQ + tok] = hv;
            }
        }
    }
}

// ---------------------------------------------------------------------------
// Kernel 3: flash attention. 1 wave (32 threads) per 16 query rows per head.
// K/V tiles are double-buffered in LDS via global_load_async_to_lds_b128
// (ASYNCcnt): issue tile j+1 while WMMAs consume tile j.
// ---------------------------------------------------------------------------
__global__ __launch_bounds__(32) void attn_kernel(
    const _Float16* __restrict__ Qh, const _Float16* __restrict__ Kh,
    const _Float16* __restrict__ Vt, float* __restrict__ out) {

    // Double-buffered operand tiles (async-staged)
    __shared__ __align__(32) _Float16 Kbuf[2][32][DHEAD];  // [key][d]   8 KB
    __shared__ __align__(32) _Float16 Vbuf[2][DHEAD][32];  // [d][key]   8 KB
    // Softmax staging
    __shared__ __align__(32) float    Sbuf[16][32];
    __shared__ __align__(32) _Float16 Pbuf[16][32];
    __shared__ float rfac[16];
    __shared__ float linv[16];

    const int lane = threadIdx.x;
    const int lm   = lane & 15;
    const int half = lane >> 4;
    const int bh = blockIdx.y;
    const int b  = bh / HEADS;
    const int h  = bh % HEADS;
    const int q0 = blockIdx.x * 16;

    const _Float16* Qbase = Qh + ((b*HEADS + h) * SEQ) * DHEAD;
    const _Float16* Kbase = Kh + ((b*HEADS + h) * SEQ) * DHEAD;
    const _Float16* Vbase = Vt + ((b*HEADS + h) * DHEAD) * SEQ;

    // ---- async stage of one 32-key K tile (4 KB contiguous in global) ----
    auto load_k_tile = [&](int key0, int bufi) {
        const char* gp = (const char*)(Kbase + key0 * DHEAD);
        unsigned base = (unsigned)(unsigned long long)(uintptr_t)&Kbuf[bufi][0][0];
#pragma unroll
        for (int t = 0; t < 8; ++t) {
            unsigned off = (unsigned)(t * 512 + lane * 16);
            async_copy_b128(base + off, gp + off);
        }
    };
    // ---- async stage of one 32-key V tile (64 rows x 64B from Vt) ----
    auto load_v_tile = [&](int key0, int bufi) {
        unsigned base = (unsigned)(unsigned long long)(uintptr_t)&Vbuf[bufi][0][0];
#pragma unroll
        for (int t = 0; t < 8; ++t) {
            int flat = t * 32 + lane;          // 0..255 16B segments
            int row  = flat >> 2;              // d row  0..63
            int seg  = flat & 3;               // 8-half segment in row
            const _Float16* gp = Vbase + row * SEQ + key0 + seg * 8;
            async_copy_b128(base + (unsigned)((row * 32 + seg * 8) * 2), gp);
        }
    };

    // Q tile 16x64 in A layout: two K-chunks of 32
    v16h aq[2];
    {
        const _Float16* qp = Qbase + (q0 + lm) * DHEAD;
#pragma unroll
        for (int c = 0; c < 2; ++c) {
            h8 lo = *(const h8*)(qp + c*32 + half*16);
            h8 hi = *(const h8*)(qp + c*32 + half*16 + 8);
            aq[c] = combine16(lo, hi);
        }
    }

    // Prologue: stage tile 0
    load_k_tile(0, 0);
    load_v_tile(0, 0);

    v8f o[4] = {};
    float mrow = -INFINITY;   // valid in lanes 0-15
    float lrow = 0.0f;
    int buf = 0;

    for (int j0 = 0; j0 < SEQ; j0 += 32) {
        // Issue next tile into the other buffer, then wait for current tile
        // (16 transfers just issued may remain outstanding; earlier ones, in
        //  order, are complete once ASYNCcnt <= 16).
        if (j0 + 32 < SEQ) {
            load_k_tile(j0 + 32, buf ^ 1);
            load_v_tile(j0 + 32, buf ^ 1);
            asm volatile("s_wait_asynccnt 0x10" ::: "memory");
        } else {
            asm volatile("s_wait_asynccnt 0x0" ::: "memory");
        }

        // ---- scores S = (Q*K^T)*scale over 32 keys -> LDS (C layout) ----
#pragma unroll
        for (int jt = 0; jt < 2; ++jt) {
            v8f s = {};
#pragma unroll
            for (int c = 0; c < 2; ++c) {
                const _Float16* kp = &Kbuf[buf][jt*16 + lm][c*32 + half*16];
                v16h bk = *(const v16h*)kp;
                s = __builtin_amdgcn_wmma_f32_16x16x32_f16(
                    false, aq[c], false, bk, (short)0, s, false, false);
            }
#pragma unroll
            for (int i = 0; i < 8; ++i)
                Sbuf[half*8 + i][jt*16 + lm] = s[i] * SCALE;
        }
        __syncthreads();

        // ---- online softmax: lanes 0-15 own one row each ----
        if (lane < 16) {
            const int r = lane;
            float cmax = -INFINITY;
#pragma unroll
            for (int c = 0; c < 32; ++c) cmax = fmaxf(cmax, Sbuf[r][c]);
            const float mnew = fmaxf(mrow, cmax);
            const float rf = __expf(mrow - mnew);
            float addl = 0.0f;
#pragma unroll
            for (int c = 0; c < 32; ++c) {
                const float p = __expf(Sbuf[r][c] - mnew);
                addl += p;
                Pbuf[r][c] = (_Float16)p;
            }
            lrow = lrow * rf + addl;
            mrow = mnew;
            rfac[r] = rf;
        }
        __syncthreads();

        // ---- rescale O accumulators ----
#pragma unroll
        for (int i = 0; i < 8; ++i) {
            const float rf = rfac[half*8 + i];
#pragma unroll
            for (int t = 0; t < 4; ++t) o[t][i] *= rf;
        }

        // ---- P (A layout from LDS) x V (B operand from LDS Vbuf) ----
        v16h ap;
        {
            h8 lo = *(const h8*)&Pbuf[lm][half*16];
            h8 hi = *(const h8*)&Pbuf[lm][half*16 + 8];
            ap = combine16(lo, hi);
        }
#pragma unroll
        for (int t = 0; t < 4; ++t) {
            const _Float16* vp = &Vbuf[buf][t*16 + lm][half*16];
            v16h bv = *(const v16h*)vp;
            o[t] = __builtin_amdgcn_wmma_f32_16x16x32_f16(
                false, ap, false, bv, (short)0, o[t], false, false);
        }
        __syncthreads();
        buf ^= 1;
    }

    // ---- final normalization + f32 store to [b, n, dim] ----
    if (lane < 16) linv[lane] = 1.0f / lrow;
    __syncthreads();

    float* outbase = out + (size_t)b * SEQ * DIM + h * DHEAD;
#pragma unroll
    for (int i = 0; i < 8; ++i) {
        const int r = half*8 + i;
        const int tok = q0 + r;
        const float inv = linv[r];
#pragma unroll
        for (int t = 0; t < 4; ++t)
            outbase[(size_t)tok * DIM + t*16 + lm] = o[t][i] * inv;
    }
}

// ---------------------------------------------------------------------------
extern "C" void kernel_launch(void* const* d_in, const int* in_sizes, int n_in,
                              void* d_out, int out_size, void* d_ws, size_t ws_size,
                              hipStream_t stream) {
    const float* q  = (const float*)d_in[0];
    const float* k  = (const float*)d_in[1];
    const float* v  = (const float*)d_in[2];
    const float* Wq = (const float*)d_in[3];
    const float* bq = (const float*)d_in[4];
    const float* Wk = (const float*)d_in[5];
    const float* bk = (const float*)d_in[6];
    const float* Wv = (const float*)d_in[7];
    const float* bv = (const float*)d_in[8];

    _Float16* ws  = (_Float16*)d_ws;
    _Float16* Wqh = ws;                         // DIM*DIM halves
    _Float16* Wkh = Wqh + DIM*DIM;
    _Float16* Wvh = Wkh + DIM*DIM;
    _Float16* Qh  = Wvh + DIM*DIM;              // BATCH*SEQ*DIM halves each
    _Float16* Kh  = Qh  + MROWS*DIM;
    _Float16* Vt  = Kh  + MROWS*DIM;

    // 1) weights -> f16
    cvt_w_kernel<<<(3*DIM*DIM)/256, 256, 0, stream>>>(Wq, Wk, Wv, Wqh);

    // 2) fused projections (z selects Q/K/V), 64x64 block / 32x32 per wave
    dim3 pg(MROWS/64, DIM/64, 3);
    proj_kernel<<<pg, 128, 0, stream>>>(q, k, v, Wqh, Wkh, Wvh,
                                        bq, bk, bv, Qh, Kh, Vt);

    // 3) flash attention: one wave per 16 query rows per (b,h)
    dim3 ag(SEQ/16, BATCH*HEADS);
    attn_kernel<<<ag, 32, 0, stream>>>(Qh, Kh, Vt, (float*)d_out);
}
